// GraphFeatureAggregator_78048145703101
// MI455X (gfx1250) — matrix operand order repaired and checked by reading, not compiled, not run on hardware
//
#include <hip/hip_runtime.h>

typedef unsigned short u16;
typedef unsigned int   u32;
typedef unsigned long long u64;

typedef __attribute__((ext_vector_type(16))) __bf16 v16bf;
typedef __attribute__((ext_vector_type(8)))  float  v8f;

#define NROWS   8192
#define DDIM    512
#define TK      32
#define WG_ROWS 128
#define NTILES  (NROWS / TK)

// ---------- helpers ----------
__device__ __forceinline__ u16 f2bf(float f) {
  u32 u = __builtin_bit_cast(u32, f);
  u32 r = (u + 0x7FFFu + ((u >> 16) & 1u)) >> 16;   // RNE
  return (u16)r;
}

template<int XM>
__device__ __forceinline__ float swz(float x) {
  int r = __builtin_amdgcn_ds_swizzle(__builtin_bit_cast(int, x), (XM << 10) | 31);
  return __builtin_bit_cast(float, r);
}
__device__ __forceinline__ float redmax16(float v) {
  v = fmaxf(v, swz<1>(v)); v = fmaxf(v, swz<2>(v));
  v = fmaxf(v, swz<4>(v)); v = fmaxf(v, swz<8>(v));
  return v;
}
__device__ __forceinline__ float redsum16(float v) {
  v += swz<1>(v); v += swz<2>(v); v += swz<4>(v); v += swz<8>(v);
  return v;
}

union Frag { uint4 q[2]; v16bf v; };

// ---------- kernel 1: f32 -> bf16 convert ----------
__global__ void GFA_cvt_kernel(const float* __restrict__ X, u16* __restrict__ Xbf) {
  u32 gid = blockIdx.x * 256u + threadIdx.x;          // 2048 blocks * 256 * 8 elems
  const float4* X4 = (const float4*)X;
  float4 a = X4[gid * 2u];
  float4 b = X4[gid * 2u + 1u];
  alignas(16) u16 o[8] = { f2bf(a.x), f2bf(a.y), f2bf(a.z), f2bf(a.w),
                           f2bf(b.x), f2bf(b.y), f2bf(b.z), f2bf(b.w) };
  *(uint4*)&Xbf[(size_t)gid * 8u] = *(const uint4*)o;
}

// shared memory: ping-pong K/V tiles; Wsum overlays the (then-dead) tile buffers
union GFA_Shared {
  struct {
    u16 Krow[2][TK][520];     // row-major key tile, padded   (2 x 33.3 KB)
    u16 Vcol[2][DDIM][40];    // transposed tile, padded      (2 x 41.0 KB)
    u16 Pbuf[8][16][40];      // per-wave P staging           (10.2 KB)
  } s;
  float Wsum[8][DDIM];        // per-wave output partials (used after loop only)
};

// ---------- kernel 2: fused flash attention + mean-pool partials ----------
__global__ void __launch_bounds__(256, 2)
GFA_attn_kernel(const u16* __restrict__ Xbf, float* __restrict__ part) {
  __shared__ __align__(16) GFA_Shared sh;

  const u32 tid  = threadIdx.x;
  const u32 wv   = tid >> 5, lane = tid & 31;
  const u32 lh   = lane & 15, hi = lane >> 4;
  const u64 xbase = (u64)Xbf;

  // ---- preload Q A-fragments (16 rows x 512, bf16) into registers ----
  const u32 qrow = blockIdx.x * WG_ROWS + wv * 16u + lh;
  const uint4* Xq = (const uint4*)(Xbf + (size_t)qrow * DDIM);
  Frag Qa[16];
#pragma unroll
  for (int c = 0; c < 16; ++c) {
    u32 off = (u32)c * 32u + hi * 16u;              // A k order used here (B matches): K = {0-15} / {16-31}
    Qa[c].q[0] = Xq[off >> 3];
    Qa[c].q[1] = Xq[(off + 8u) >> 3];
  }

  v8f O[32];
#pragma unroll
  for (int t = 0; t < 32; ++t) O[t] = (v8f){0.f,0.f,0.f,0.f,0.f,0.f,0.f,0.f};
  float m[8], l[8];
#pragma unroll
  for (int r = 0; r < 8; ++r) { m[r] = -3.0e38f; l[r] = 0.0f; }

  // per-thread copy/transpose coordinates (2048 chunks of 8 bf16 over 32x512 tile)
  const u32 crow0 = tid >> 6, c8 = tid & 63;        // + i*4 rows per step (256 thr * 8 it)

  // ================= prologue: fetch + build tile 0 into buffer 0 =================
#pragma unroll
  for (int i = 0; i < 8; ++i) {
    u32 row = crow0 + (u32)i * 4u;
    u32 voff  = (row * DDIM + c8 * 8u) * 2u;        // kb = 0
    u32 laddr = (u32)(u64)&sh.s.Krow[0][row][c8 * 8u];
    asm volatile("global_load_async_to_lds_b128 %0, %1, %2"
                 :: "v"(laddr), "v"(voff), "s"(xbase) : "memory");
  }
#if __has_builtin(__builtin_amdgcn_s_wait_asynccnt)
  __builtin_amdgcn_s_wait_asynccnt(0);
#else
  asm volatile("s_wait_asynccnt 0" ::: "memory");
#endif
  __syncthreads();
#pragma unroll
  for (int i = 0; i < 8; ++i) {
    u32 row = crow0 + (u32)i * 4u;
    union { uint4 q; u16 e[8]; } w;
    w.q = *(const uint4*)&sh.s.Krow[0][row][c8 * 8u];
#pragma unroll
    for (int j = 0; j < 8; ++j) sh.s.Vcol[0][c8 * 8u + j][row] = w.e[j];
  }
  __syncthreads();

  // ================= main loop (double-buffered) =================
  u32 p = 0;
#pragma unroll 1
  for (u32 kt = 0; kt < NTILES; ++kt) {
    const u32 q = p ^ 1u;
    const bool more = (kt + 1u) < NTILES;

    // ---- issue async copy of NEXT tile into buffer q (overlaps with compute) ----
    if (more) {
      const u32 kb2 = (kt + 1u) * TK;
#pragma unroll
      for (int i = 0; i < 8; ++i) {
        u32 row = crow0 + (u32)i * 4u;
        u32 voff  = ((kb2 + row) * DDIM + c8 * 8u) * 2u;
        u32 laddr = (u32)(u64)&sh.s.Krow[q][row][c8 * 8u];
        asm volatile("global_load_async_to_lds_b128 %0, %1, %2"
                     :: "v"(laddr), "v"(voff), "s"(xbase) : "memory");
      }
    }

    // ---- S = Q K^T : two 16x16 f32 tiles over 32 keys, contract D in 16 chunks ----
    const u16* krp = &sh.s.Krow[p][0][0];
    v8f s0 = (v8f){0.f,0.f,0.f,0.f,0.f,0.f,0.f,0.f};
    v8f s1 = (v8f){0.f,0.f,0.f,0.f,0.f,0.f,0.f,0.f};
#pragma unroll
    for (int c = 0; c < 16; ++c) {
      const u16* kp0 = krp + lh * 520u + (u32)c * 32u + hi * 16u;          // keys 0..15
      const u16* kp1 = krp + (lh + 16u) * 520u + (u32)c * 32u + hi * 16u;  // keys 16..31
      Frag b0, b1;
      b0.q[0] = *(const uint4*)kp0;  b0.q[1] = *(const uint4*)(kp0 + 8);
      b1.q[0] = *(const uint4*)kp1;  b1.q[1] = *(const uint4*)(kp1 + 8);
      s0 = __builtin_amdgcn_wmma_f32_16x16x32_bf16(false, Qa[c].v, false, b0.v, (short)0, s0, false, false);
      s1 = __builtin_amdgcn_wmma_f32_16x16x32_bf16(false, Qa[c].v, false, b1.v, (short)0, s1, false, false);
    }

    // ---- online softmax update (f32), stage P (bf16) into per-wave LDS ----
    float alpha[8];
#pragma unroll
    for (int r = 0; r < 8; ++r) {
      float tm = redmax16(fmaxf(s0[r], s1[r]));    // row max over 32 keys
      float mn = fmaxf(m[r], tm);
      alpha[r] = __expf(m[r] - mn);
      float p0 = __expf(s0[r] - mn);
      float p1 = __expf(s1[r] - mn);
      l[r] = l[r] * alpha[r] + redsum16(p0 + p1);
      m[r] = mn;
      u32 rowi = (u32)r + hi * 8u;                 // C layout: lanes>=16 hold rows 8..15
      sh.s.Pbuf[wv][rowi][lh]       = f2bf(p0);
      sh.s.Pbuf[wv][rowi][16u + lh] = f2bf(p1);
    }
    asm volatile("" ::: "memory");                 // keep P stores before P reloads (DS in-order)

    // rescale accumulator
#pragma unroll
    for (int t = 0; t < 32; ++t)
#pragma unroll
      for (int r = 0; r < 8; ++r) O[t][r] *= alpha[r];

    // ---- reload P as A-fragment ----
    Frag Pa;
    const u16* pp = &sh.s.Pbuf[wv][lh][hi * 16u];
    Pa.q[0] = *(const uint4*)pp;
    Pa.q[1] = *(const uint4*)(pp + 8);

    // ---- O += P V : 32 d-tiles, B-fragments from transposed LDS ----
    const u16* vcp = &sh.s.Vcol[p][0][0];
#pragma unroll
    for (int t = 0; t < 32; ++t) {
      const u16* vp = vcp + ((u32)t * 16u + lh) * 40u + hi * 16u;
      Frag bv;
      bv.q[0] = *(const uint4*)vp;
      bv.q[1] = *(const uint4*)(vp + 8);
      O[t] = __builtin_amdgcn_wmma_f32_16x16x32_bf16(false, Pa.v, false, bv.v, (short)0, O[t], false, false);
    }

    // ---- retire async copy; build transposed layout for next tile ----
    if (more) {
#if __has_builtin(__builtin_amdgcn_s_wait_asynccnt)
      __builtin_amdgcn_s_wait_asynccnt(0);
#else
      asm volatile("s_wait_asynccnt 0" ::: "memory");
#endif
    }
    __syncthreads();                               // compute on buf p done; async into buf q landed
    if (more) {
#pragma unroll
      for (int i = 0; i < 8; ++i) {
        u32 row = crow0 + (u32)i * 4u;
        union { uint4 q4; u16 e[8]; } w;
        w.q4 = *(const uint4*)&sh.s.Krow[q][row][c8 * 8u];
#pragma unroll
        for (int j = 0; j < 8; ++j) sh.s.Vcol[q][c8 * 8u + j][row] = w.e[j];
      }
    }
    __syncthreads();                               // Vcol[q] visible; buf p free for reuse
    p = q;
  }

  // ---- finalize: (O/l) summed over this wave's 16 rows -> Wsum[wv][d] ----
  float inv[8];
#pragma unroll
  for (int r = 0; r < 8; ++r) inv[r] = 1.0f / l[r];
#pragma unroll
  for (int t = 0; t < 32; ++t) {
    float s = 0.0f;
#pragma unroll
    for (int r = 0; r < 8; ++r) s += O[t][r] * inv[r];
    s += swz<16>(s);                               // combine row-halves (same d in both halves)
    if (hi == 0) sh.Wsum[wv][(u32)t * 16u + lh] = s;
  }
  __syncthreads();

  // deterministic fixed-order reduction over the 8 waves
  for (u32 d = tid; d < DDIM; d += 256u) {
    float s = 0.0f;
#pragma unroll
    for (int w = 0; w < 8; ++w) s += sh.Wsum[w][d];
    part[(size_t)blockIdx.x * DDIM + d] = s;
  }
}

// ---------- kernel 3: sum 64 block partials, scale by 1/N ----------
__global__ void GFA_fin_kernel(const float* __restrict__ part, float* __restrict__ out) {
  u32 d = blockIdx.x * 256u + threadIdx.x;         // 512 threads total
  float s = 0.0f;
  for (int b = 0; b < NROWS / WG_ROWS; ++b) s += part[(size_t)b * DDIM + d];
  out[d] = s * (1.0f / (float)NROWS);
}

extern "C" void kernel_launch(void* const* d_in, const int* in_sizes, int n_in,
                              void* d_out, int out_size, void* d_ws, size_t ws_size,
                              hipStream_t stream) {
  const float* X = (const float*)d_in[0];
  float* out = (float*)d_out;
  u16*   Xbf  = (u16*)d_ws;                                          // 8 MB
  float* part = (float*)((char*)d_ws + (size_t)NROWS * DDIM * 2);    // 64*512 f32

  GFA_cvt_kernel<<<(NROWS * DDIM / 8) / 256, 256, 0, stream>>>(X, Xbf);
  GFA_attn_kernel<<<NROWS / WG_ROWS, 256, 0, stream>>>(Xbf, part);
  GFA_fin_kernel<<<2, 256, 0, stream>>>(part, out);
}
